// SigmaMoELayer_19404662243921
// MI455X (gfx1250) — matrix operand order repaired and checked by reading, not compiled, not run on hardware
//
#include <hip/hip_runtime.h>
#include <hip/hip_bf16.h>
#include <cstdint>
#include <cstddef>

// Problem constants (from reference): B=4,S=2048 -> T=8192 tokens
#define T_TOK 8192
#define DMODEL 1024
#define NEXP 32
#define HDIM 128

typedef __attribute__((ext_vector_type(16))) __bf16 v16bf;
typedef __attribute__((ext_vector_type(8)))  __bf16 v8bf;
typedef __attribute__((ext_vector_type(4)))  __bf16 v4bf;
typedef __attribute__((ext_vector_type(8)))  float  v8f;

typedef __attribute__((ext_vector_type(4))) unsigned int tdm_v4u;
typedef __attribute__((ext_vector_type(8))) int          tdm_v8i;
typedef __attribute__((ext_vector_type(4))) int          tdm_v4i;

// ---------------------------------------------------------------------------
// WMMA fragment helpers (16x16x32 bf16, wave32) — this code's contiguous K order, same for A and B.
// A (16x32 MxK): lane l -> row M=l%16, K = 16*(l/16) + j; 16 contiguous K elements.
// B (32x16 KxN): lane l -> col N=l%16, K-half l/16; j = K%16 -> 16 contiguous
//   K elements; weights pre-swizzled so a B fragment is 32 contiguous bytes.
// ---------------------------------------------------------------------------
static __device__ __forceinline__ v16bf combine16(v8bf lo, v8bf hi) {
  return __builtin_shufflevector(lo, hi, 0,1,2,3,4,5,6,7,8,9,10,11,12,13,14,15);
}
static __device__ __forceinline__ v16bf load_a_frag(const __bf16* prow, int k32, int kh) {
  const __bf16* p = prow + k32*32 + kh*16;
  v8bf lo = *(const v8bf*)p;
  v8bf hi = *(const v8bf*)(p + 8);
  return combine16(lo, hi);
}
static __device__ __forceinline__ v16bf load_b_frag(const __bf16* p) {
  v8bf lo = *(const v8bf*)p;
  v8bf hi = *(const v8bf*)(p + 8);
  return combine16(lo, hi);
}
static __device__ __forceinline__ v8f wmma_bf16(v16bf a, v16bf b, v8f c) {
  return __builtin_amdgcn_wmma_f32_16x16x32_bf16(false, a, false, b, (short)0, c, false, false);
}

// ---------------------------------------------------------------------------
// Tensor Data Mover: async 1D copy global -> LDS (contiguous block).
// D# per ISA 8.3/8.4: group0 = {count=1 | lds_addr | global_addr | type=2},
// group1 = {data_size=8B, tensor_dim0=tile_dim0=n8, tile_dim1=0 (1D)}.
// Tracked by TENSORcnt; wave issues it and continues (true async).
// ---------------------------------------------------------------------------
static __device__ __forceinline__ void tdm_load_1d(uint32_t lds_off,
                                                   const void* gptr,
                                                   uint32_t n8 /* 8-byte units */) {
  uint64_t ga = (uint64_t)(uintptr_t)gptr;
  tdm_v4u g0;
  g0.x = 1u;                                                   // count=1 (valid user D#)
  g0.y = lds_off;                                              // lds_addr (bytes)
  g0.z = (uint32_t)ga;                                         // global_addr[31:0]
  g0.w = (uint32_t)((ga >> 32) & 0x1FFFFFFull) | (2u << 30);   // ga[56:32] | type=2
  tdm_v8i g1;
  g1[0] = (int)(3u << 16);                      // workgroup_mask=0, data_size=8B
  g1[1] = (int)((n8 & 0xFFFFu) << 16);          // tensor_dim0[15:0] << 16
  g1[2] = (int)(((n8 >> 16) & 0xFFFFu) | (1u << 16)); // dim0[31:16] | tensor_dim1=1
  g1[3] = (int)((n8 & 0xFFFFu) << 16);          // tensor_dim1[31:16]=0 | tile_dim0=n8
  g1[4] = 0;                                    // tile_dim1=0, tile_dim2=0 -> 1D
  g1[5] = (int)n8;                              // tensor_dim0_stride[31:0]
  g1[6] = 0;
  g1[7] = 0;
  tdm_v4i z4 = {0, 0, 0, 0};
#if defined(__clang_major__) && __clang_major__ >= 23
  tdm_v8i z8 = {0, 0, 0, 0, 0, 0, 0, 0};
  __builtin_amdgcn_tensor_load_to_lds(g0, g1, z4, z4, z8, 0);
#else
  __builtin_amdgcn_tensor_load_to_lds(g0, g1, z4, z4, 0);
#endif
}

// ---------------------------------------------------------------------------
// x: f32 -> bf16, row-major [T, D]
// ---------------------------------------------------------------------------
__global__ __launch_bounds__(256) void cvt_x_kernel(const float* __restrict__ x,
                                                    __bf16* __restrict__ xb, int n4) {
  int i = blockIdx.x * 256 + threadIdx.x;
  if (i >= n4) return;
  float4 v = ((const float4*)x)[i];
  v4bf o;
  o.x = (__bf16)v.x; o.y = (__bf16)v.y; o.z = (__bf16)v.z; o.w = (__bf16)v.w;
  ((v4bf*)xb)[i] = o;
}

// ---------------------------------------------------------------------------
// Weight swizzle into B-fragment layout.
//   korder=0: idx2 = ntile*K32 + k32  (N-chunks contiguous; for w_values/w_sel)
//   korder=1: idx2 = k32*NT + ntile   (K-chunks contiguous; for w_keys)
// dst[e][(idx2*32 + lane)*16 + j], n = ntile*16 + lane%16,
// k = k32*32 + 16*(lane/16) + j.  trans=1: src is [N,K] (router weight).
// ---------------------------------------------------------------------------
__global__ __launch_bounds__(256) void swizzle_b_kernel(const float* __restrict__ src,
                                                        __bf16* __restrict__ dst,
                                                        int nExp, int Kdim, int Ndim,
                                                        int trans, int korder) {
  size_t total = (size_t)nExp * Kdim * Ndim;
  size_t i = (size_t)blockIdx.x * 256 + threadIdx.x;
  if (i >= total) return;
  size_t perE = (size_t)Kdim * Ndim;
  int e   = (int)(i / perE);
  int rem = (int)(i % perE);
  int j    = rem & 15;
  int lane = (rem >> 4) & 31;
  int K32  = Kdim >> 5;
  int NT   = Ndim >> 4;
  int idx2 = rem >> 9;
  int k32, ntile;
  if (korder) { ntile = idx2 % NT; k32 = idx2 / NT; }
  else        { k32 = idx2 % K32; ntile = idx2 / K32; }
  int n = ntile * 16 + (lane & 15);
  int k = k32 * 32 + ((lane >> 4) << 4) + j;
  size_t si = trans ? ((size_t)n * Kdim + k)
                    : (((size_t)e * Kdim + k) * Ndim + n);
  dst[i] = (__bf16)src[si];
}

// ---------------------------------------------------------------------------
// Router: logits = X @ w_sel^T via WMMA (128-token tile, 8 waves x 16 rows),
// per-row top-2 + sigmoid gate, atomic append into per-expert lists.
// ---------------------------------------------------------------------------
__global__ __launch_bounds__(256) void moe_router_kernel(
    const __bf16* __restrict__ xb, const __bf16* __restrict__ wsel,
    int* __restrict__ cnt, int* __restrict__ tokl, float* __restrict__ gatel) {
  __shared__ __align__(16) float sel[8 * 16 * 32];
  int tid = threadIdx.x, wave = tid >> 5, lane = tid & 31;
  int kh = lane >> 4, l16 = lane & 15;
  int m0 = blockIdx.x * 128;
  int rowA = m0 + wave * 16 + l16;
  const __bf16* prow = xb + (size_t)rowA * DMODEL;

  v8f z = {0.f,0.f,0.f,0.f,0.f,0.f,0.f,0.f};
  v8f acc0 = z, acc1 = z;
  for (int k32 = 0; k32 < DMODEL / 32; ++k32) {
    v16bf a  = load_a_frag(prow, k32, kh);
    v16bf b0 = load_b_frag(wsel + ((size_t)((0 * 32 + k32) * 32 + lane)) * 16);
    v16bf b1 = load_b_frag(wsel + ((size_t)((1 * 32 + k32) * 32 + lane)) * 16);
    acc0 = wmma_bf16(a, b0, acc0);
    acc1 = wmma_bf16(a, b1, acc1);
  }
  float* sl = sel + wave * 512;
#pragma unroll
  for (int r = 0; r < 8; ++r) {
    int m = 8 * kh + r;
    sl[m * 32 + l16]      = acc0[r];
    sl[m * 32 + 16 + l16] = acc1[r];
  }
  __syncthreads();
  if (lane < 16) {
    const float* rowp = sl + lane * 32;
    float b1v = -1e30f, b2v = -1e30f; int i1 = 0, i2 = 0;
#pragma unroll
    for (int e = 0; e < NEXP; ++e) {
      float v = rowp[e];
      if (v > b1v) { b2v = b1v; i2 = i1; b1v = v; i1 = e; }
      else if (v > b2v) { b2v = v; i2 = e; }
    }
    int t = m0 + wave * 16 + lane;
    float g1 = 1.f / (1.f + __expf(-b1v));
    float g2 = 1.f / (1.f + __expf(-b2v));
    int p1 = atomicAdd(&cnt[i1], 1);
    tokl[i1 * T_TOK + p1] = t; gatel[i1 * T_TOK + p1] = g1;
    int p2 = atomicAdd(&cnt[i2], 1);
    tokl[i2 * T_TOK + p2] = t; gatel[i2 * T_TOK + p2] = g2;
  }
}

// ---------------------------------------------------------------------------
// Expert kernel: block = (expert e, 128-token tile of e's gathered list).
// B operands TDM-staged into double-buffered LDS (32KB chunks, wave 0 issues,
// all 8 waves consume) -> 8x less L2 read traffic than per-wave streaming.
// GEMM1: H = relu(X @ Wk[e]) * gate -> per-wave LDS strip (bf16).
// GEMM2: H @ Wv[e], scatter-accumulate with f32 global atomics.
// ---------------------------------------------------------------------------
__global__ __launch_bounds__(256) void moe_expert_kernel(
    const __bf16* __restrict__ xb, const __bf16* __restrict__ wk_all,
    const __bf16* __restrict__ wv_all, const int* __restrict__ cnt,
    const int* __restrict__ tokl, const float* __restrict__ gatel,
    float* __restrict__ out) {
  __shared__ __align__(16) __bf16 Hlds[8 * 16 * HDIM];   // 32 KB static
  extern __shared__ __align__(16) __bf16 Bst[];          // 2 x 16384 bf16 (64 KB dynamic)
  __bf16* Bbuf0 = Bst;
  __bf16* Bbuf1 = Bst + 16384;

  int e  = blockIdx.x >> 6;
  int m0 = (blockIdx.x & 63) * 128;
  int n_tok = cnt[e];
  if (m0 >= n_tok) return;  // uniform per block

  int tid = threadIdx.x, wave = tid >> 5, lane = tid & 31;
  int kh = lane >> 4, l16 = lane & 15;
  const __bf16* wk = wk_all + (size_t)e * DMODEL * HDIM;  // k-outer swizzle: 8 x 32KB chunks
  const __bf16* wv = wv_all + (size_t)e * HDIM * DMODEL;  // n-outer swizzle: 8 x 32KB chunks

  // Kick off first GEMM1 B chunk before doing anything else (async DMA).
  if (wave == 0)
    tdm_load_1d((uint32_t)(uintptr_t)Bbuf0, wk, 4096);

  const int*   tl = tokl + e * T_TOK;
  const float* gl = gatel + e * T_TOK;

  int gA = m0 + wave * 16 + l16;          // A-fragment row (pads clamp, gate=0)
  if (gA > n_tok - 1) gA = n_tok - 1;
  int tA = tl[gA];

  int tokR[8]; float g[8];                // C-row tokens & gates: M = base+8*kh+r
  int basem = m0 + wave * 16 + 8 * kh;
#pragma unroll
  for (int r = 0; r < 8; ++r) {
    int gi = basem + r;
    if (gi < n_tok) { tokR[r] = tl[gi]; g[r] = gl[gi]; }
    else            { tokR[r] = tl[n_tok - 1]; g[r] = 0.f; }
  }

  const __bf16* prow = xb + (size_t)tA * DMODEL;
  v8f z = {0.f,0.f,0.f,0.f,0.f,0.f,0.f,0.f};

  // ---- GEMM1: [16 x D] @ [D x 128] per wave; B from TDM-staged LDS ----
  v8f acc[8];
#pragma unroll
  for (int n = 0; n < 8; ++n) acc[n] = z;
  for (int kc = 0; kc < 8; ++kc) {
    if (wave == 0) __builtin_amdgcn_s_wait_tensorcnt(0);  // chunk kc landed
    __syncthreads();                                      // publish + retire old buf
    if (wave == 0 && kc + 1 < 8) {
      __bf16* nb = ((kc + 1) & 1) ? Bbuf1 : Bbuf0;
      tdm_load_1d((uint32_t)(uintptr_t)nb, wk + (size_t)(kc + 1) * 16384, 4096);
    }
    const __bf16* bs = (kc & 1) ? Bbuf1 : Bbuf0;
#pragma unroll
    for (int q = 0; q < 4; ++q) {                         // k32 = kc*4 + q
      v16bf a = load_a_frag(prow, kc * 4 + q, kh);
#pragma unroll
      for (int n = 0; n < 8; ++n) {
        v16bf b = load_b_frag(bs + ((size_t)((q * 8 + n) * 32 + lane)) * 16);
        acc[n] = wmma_bf16(a, b, acc[n]);
      }
    }
  }

  // Prefetch first GEMM2 B chunk; overlaps with H-tile epilogue below.
  if (wave == 0)
    tdm_load_1d((uint32_t)(uintptr_t)Bbuf0, wv, 4096);

  // relu * gate -> per-wave LDS strip (row-major [16][128] bf16)
  __bf16* hl = Hlds + wave * (16 * HDIM);
#pragma unroll
  for (int n = 0; n < 8; ++n) {
#pragma unroll
    for (int r = 0; r < 8; ++r) {
      float v = acc[n][r];
      v = (v > 0.f ? v : 0.f) * g[r];
      hl[(8 * kh + r) * HDIM + n * 16 + l16] = (__bf16)v;
    }
  }

  // ---- GEMM2: [16 x 128] @ [128 x 1024] per wave; B from TDM-staged LDS ----
  for (int nc = 0; nc < 8; ++nc) {
    if (wave == 0) __builtin_amdgcn_s_wait_tensorcnt(0);
    __syncthreads();                                      // also publishes Hlds (nc==0)
    if (wave == 0 && nc + 1 < 8) {
      __bf16* nb = ((nc + 1) & 1) ? Bbuf1 : Bbuf0;
      tdm_load_1d((uint32_t)(uintptr_t)nb, wv + (size_t)(nc + 1) * 16384, 4096);
    }
    const __bf16* bs = (nc & 1) ? Bbuf1 : Bbuf0;

    v16bf a2[4];
#pragma unroll
    for (int k2 = 0; k2 < 4; ++k2)
      a2[k2] = load_a_frag(hl + l16 * HDIM, k2, kh);

    v8f o[8];
#pragma unroll
    for (int n = 0; n < 8; ++n) o[n] = z;
#pragma unroll
    for (int k2 = 0; k2 < 4; ++k2) {
#pragma unroll
      for (int n = 0; n < 8; ++n) {
        v16bf b = load_b_frag(bs + ((size_t)((n * 4 + k2) * 32 + lane)) * 16);
        o[n] = wmma_bf16(a2[k2], b, o[n]);
      }
    }
#pragma unroll
    for (int n = 0; n < 8; ++n) {
      int col = (nc * 8 + n) * 16 + l16;
#pragma unroll
      for (int r = 0; r < 8; ++r)
        unsafeAtomicAdd(&out[(size_t)tokR[r] * DMODEL + col], o[n][r]);
    }
  }
}

// ---------------------------------------------------------------------------
// Workspace layout (bytes):
//   [0)            xb      bf16  T*D        = 16,777,216
//   [16,777,216)   wsel_s  bf16  D*32       =     65,536
//   [16,842,752)   wk_s    bf16  E*D*H      =  8,388,608   (k-outer swizzle)
//   [25,231,360)   wv_s    bf16  E*H*D      =  8,388,608   (n-outer swizzle)
//   [33,619,968)   counts  i32   E          (padded to 256)
//   [33,620,224)   tokl    i32   E*T        =  1,048,576
//   [34,668,800)   gatel   f32   E*T        =  1,048,576
//   total ~= 35.7 MB
// ---------------------------------------------------------------------------
extern "C" void kernel_launch(void* const* d_in, const int* in_sizes, int n_in,
                              void* d_out, int out_size, void* d_ws, size_t ws_size,
                              hipStream_t stream) {
  const float* x     = (const float*)d_in[0];
  const float* wsel  = (const float*)d_in[1];
  const float* wkeys = (const float*)d_in[2];
  const float* wvals = (const float*)d_in[3];
  (void)in_sizes; (void)n_in; (void)out_size; (void)ws_size;

  char* ws = (char*)d_ws;
  __bf16* xb    = (__bf16*)(ws);
  __bf16* wselb = (__bf16*)(ws + 16777216);
  __bf16* wkb   = (__bf16*)(ws + 16842752);
  __bf16* wvb   = (__bf16*)(ws + 25231360);
  int*    cntp  = (int*)   (ws + 33619968);
  int*    tokl  = (int*)   (ws + 33620224);
  float*  gatel = (float*) (ws + 34668800);
  float*  out   = (float*)d_out;

  hipMemsetAsync(cntp, 0, NEXP * sizeof(int), stream);
  hipMemsetAsync(out, 0, (size_t)T_TOK * DMODEL * sizeof(float), stream);

  int n4 = T_TOK * DMODEL / 4;
  cvt_x_kernel<<<(n4 + 255) / 256, 256, 0, stream>>>(x, xb, n4);

  swizzle_b_kernel<<<(DMODEL * 32 + 255) / 256, 256, 0, stream>>>(
      wsel, wselb, 1, DMODEL, NEXP, 1, 0);
  swizzle_b_kernel<<<(NEXP * DMODEL * HDIM + 255) / 256, 256, 0, stream>>>(
      wkeys, wkb, NEXP, DMODEL, HDIM, 0, 1);   // k-outer: contiguous 32KB k-chunks
  swizzle_b_kernel<<<(NEXP * HDIM * DMODEL + 255) / 256, 256, 0, stream>>>(
      wvals, wvb, NEXP, HDIM, DMODEL, 0, 0);   // n-outer: contiguous 32KB n-chunks

  moe_router_kernel<<<T_TOK / 128, 256, 0, stream>>>(xb, wselb, cntp, tokl, gatel);

  moe_expert_kernel<<<NEXP * (T_TOK / 128), 256, 2 * 16384 * sizeof(__bf16), stream>>>(
      xb, wkb, wvb, cntp, tokl, gatel, out);
}
